// UncertainReFineBlock_24644522344928
// MI455X (gfx1250) — matrix operand order repaired and checked by reading, not compiled, not run on hardware
//
#include <hip/hip_runtime.h>

typedef float    v8f  __attribute__((ext_vector_type(8)));
typedef _Float16 v4h  __attribute__((ext_vector_type(4)));
typedef _Float16 v8h  __attribute__((ext_vector_type(8)));
typedef _Float16 v16h __attribute__((ext_vector_type(16)));

// ---------------------------------------------------------------------------
// Constants
// ---------------------------------------------------------------------------
constexpr int kC     = 64;   // channels everywhere
constexpr int kNLay  = 5;    // w_in, 3x ws_mid, w_out
constexpr int kSH    = 72;   // f16 row stride (36 dwords): b128 frag loads touch each bank exactly 2x
constexpr int kTX    = 16;   // conv tile width
constexpr int kTY    = 4;    // conv tile height
constexpr int kFMW   = kTX + 4;    // 20
constexpr int kFMH   = kTY + 4;    // 8
constexpr int kRW    = kTX + 2;    // 18
constexpr int kRH    = kTY + 2;    // 6
constexpr int kRPX   = kRW * kRH;  // 108 real conv1 pixels
constexpr int kRROWS = 112;        // padded to 7 WMMA M-tiles
constexpr int kOPX   = kTX * kTY;  // 64 output pixels (4 M-tiles)

__device__ __forceinline__ float relu6f(float v) { return fminf(fmaxf(v, 0.0f), 6.0f); }

// D(16x16,f32) += A(16x32,f16) x B(32x16,f16)  -- probe-confirmed signature
__device__ __forceinline__ v8f wmma32(v16h a, v16h b, v8f c) {
  return __builtin_amdgcn_wmma_f32_16x16x32_f16(false, a, false, b, (short)0, c, false, false);
}

// Assemble a 16-half fragment from two 16-byte LDS regions
__device__ __forceinline__ v16h ldfrag2(const _Float16* p0, const _Float16* p1) {
  v8h a = *(const v8h*)p0;
  v8h b = *(const v8h*)p1;
  return __builtin_shufflevector(a, b, 0, 1, 2, 3, 4, 5, 6, 7, 8, 9, 10, 11, 12, 13, 14, 15);
}

// Split fp32 -> f16 hi + f16 lo (hi+lo carries ~22 mantissa bits)
__device__ __forceinline__ void split4(float4 w, v4h& hi, v4h& lo) {
  hi.x = (_Float16)w.x; lo.x = (_Float16)(w.x - (float)hi.x);
  hi.y = (_Float16)w.y; lo.y = (_Float16)(w.y - (float)hi.y);
  hi.z = (_Float16)w.z; lo.z = (_Float16)(w.z - (float)hi.z);
  hi.w = (_Float16)w.w; lo.w = (_Float16)(w.w - (float)hi.w);
}

// ---------------------------------------------------------------------------
// Kernel 1: fused 5-layer MLP (relu6, mid-layer residuals) + mask select.
// Split-f16 WMMA: acc += Ahi*Bhi + Ahi*Blo + Alo*Bhi  (fp32 accumulate).
// 256 threads = 8 waves, 16 rows per wave.
// ---------------------------------------------------------------------------
__launch_bounds__(256, 1)
__global__ void mlp_kernel(const float* __restrict__ x, const int* __restrict__ mask,
                           const float* __restrict__ w_in, const float* __restrict__ b_in,
                           const float* __restrict__ ws_mid, const float* __restrict__ bs_mid,
                           const float* __restrict__ w_out, const float* __restrict__ b_out,
                           float* __restrict__ fm) {
  __shared__ _Float16 Whi[kNLay][kC * kSH];   // 46080 B
  __shared__ _Float16 Wlo[kNLay][kC * kSH];   // 46080 B
  __shared__ float    Bl [kNLay][kC];         //  1280 B
  __shared__ float    xs [8][16 * kC];        // 32768 B  exact x for mask passthrough
  __shared__ _Float16 uhi[8][16 * kSH];       // 18432 B  per-wave activations (hi)
  __shared__ _Float16 ulo[8][16 * kSH];       // 18432 B  per-wave activations (lo)
                                              // total ~159.3 KB

  const int tid  = threadIdx.x;
  const int wave = tid >> 5;
  const int lane = tid & 31;

  // ---- stage weights as split f16 hi/lo (L2-resident source) ----
  for (int i = tid; i < kNLay * kC * 16; i += 256) {
    int l  = i / (kC * 16);
    int r  = i - l * (kC * 16);
    int n  = r >> 4;
    int k4 = r & 15;
    const float* src = (l == 0) ? w_in : (l == 4) ? w_out : (ws_mid + (l - 1) * kC * kC);
    float4 w = *(const float4*)(src + n * kC + k4 * 4);
    v4h hi, lo;
    split4(w, hi, lo);
    *(v4h*)(&Whi[l][n * kSH + k4 * 4]) = hi;
    *(v4h*)(&Wlo[l][n * kSH + k4 * 4]) = lo;
  }
  for (int i = tid; i < kNLay * kC; i += 256) {
    int l = i >> 6, n = i & 63;
    Bl[l][n] = (l == 0) ? b_in[n] : (l == 4) ? b_out[n] : bs_mid[(l - 1) * kC + n];
  }

  // ---- stage 16 input rows per wave: exact f32 copy + split f16 hi/lo ----
  const long rowbase = (long)blockIdx.x * 128 + wave * 16;
  const float* xg = x + rowbase * kC;
  for (int i = lane; i < 16 * 16; i += 32) {
    int r = i >> 4, k4 = i & 15;
    float4 w = *(const float4*)(xg + r * kC + k4 * 4);
    *(float4*)(&xs[wave][r * kC + k4 * 4]) = w;
    v4h hi, lo;
    split4(w, hi, lo);
    *(v4h*)(&uhi[wave][r * kSH + k4 * 4]) = hi;
    *(v4h*)(&ulo[wave][r * kSH + k4 * 4]) = lo;
  }
  __syncthreads();

  const int hlf = lane >> 4;
  const int lm  = lane & 15;
  const v8f vzero = {0.f, 0.f, 0.f, 0.f, 0.f, 0.f, 0.f, 0.f};
  _Float16* uh = uhi[wave];
  _Float16* ul = ulo[wave];

  #pragma unroll
  for (int l = 0; l < kNLay; ++l) {
    v8f acc[4];
    #pragma unroll
    for (int nb = 0; nb < 4; ++nb) acc[nb] = vzero;

    #pragma unroll
    for (int kb = 0; kb < kC; kb += 32) {
      // A frag (16x32 f16): lanes<16 K 0-7/16-23, lanes>=16 K 8-15/24-31
      const _Float16* pah = uh + lm * kSH + kb + 8 * hlf;
      const _Float16* pal = ul + lm * kSH + kb + 8 * hlf;
      v16h ah = ldfrag2(pah, pah + 16);
      v16h al = ldfrag2(pal, pal + 16);
      #pragma unroll
      for (int nb = 0; nb < 4; ++nb) {
        const int n = nb * 16 + lm;
        // B frag (32x16 f16): lanes<16 K 0-7/16-23, lanes>=16 K 8-15/24-31
        const _Float16* pbh = &Whi[l][n * kSH + kb + 8 * hlf];
        const _Float16* pbl = &Wlo[l][n * kSH + kb + 8 * hlf];
        v16h bh  = ldfrag2(pbh, pbh + 16);
        v16h blo = ldfrag2(pbl, pbl + 16);
        acc[nb] = wmma32(ah, bh,  acc[nb]);
        acc[nb] = wmma32(ah, blo, acc[nb]);
        acc[nb] = wmma32(al, bh,  acc[nb]);
      }
    }

    if (l < 4) {
      const bool residual = (l >= 1);   // layers 1..3: u = u + relu6(mlp(u))
      #pragma unroll
      for (int nb = 0; nb < 4; ++nb) {
        const int n  = nb * 16 + lm;
        const float bn = Bl[l][n];
        #pragma unroll
        for (int j = 0; j < 8; ++j) {
          const int m = j + 8 * hlf;   // D layout: VGPR j, upper lanes -> M+8
          float v = relu6f(acc[nb][j] + bn);
          if (residual) v += (float)uh[m * kSH + n] + (float)ul[m * kSH + n];
          _Float16 hi = (_Float16)v;
          uh[m * kSH + n] = hi;                       // same-wave DS stays in order
          ul[m * kSH + n] = (_Float16)(v - (float)hi);
        }
      }
    } else {
      // final layer: relu6 + mask select + fused global store
      #pragma unroll
      for (int nb = 0; nb < 4; ++nb) {
        const int n  = nb * 16 + lm;
        const float bn = Bl[4][n];
        #pragma unroll
        for (int j = 0; j < 8; ++j) {
          const int m = j + 8 * hlf;
          float v = relu6f(acc[nb][j] + bn);
          float o = (mask[rowbase + m] != 0) ? v : xs[wave][m * kC + n];
          fm[(rowbase + m) * kC + n] = o;
        }
      }
    }
  }
}

// ---------------------------------------------------------------------------
// Kernel 2: fused (fm + sepconv1) -> sepconv2 -> out. One 16x4 output tile per
// workgroup; depthwise in fp32 VALU, pointwise via split-f16 WMMA.
// ---------------------------------------------------------------------------
__launch_bounds__(256, 1)
__global__ void conv_kernel(const float* __restrict__ fm,
                            const float* __restrict__ dw1, const float* __restrict__ pw1,
                            const float* __restrict__ g1,  const float* __restrict__ be1,
                            const float* __restrict__ dw2, const float* __restrict__ pw2,
                            const float* __restrict__ g2,  const float* __restrict__ be2,
                            float* __restrict__ out, int H, int W) {
  __shared__ float    fm_s[kFMH * kFMW * kC];   // 40960 B (later reused for t2 hi/lo)
  __shared__ _Float16 t_hi[kRROWS * kSH];       // 16128 B depthwise1 out (hi)
  __shared__ _Float16 t_lo[kRROWS * kSH];       // 16128 B depthwise1 out (lo)
  __shared__ float    r_s [kRROWS * kC];        // 28672 B conv1 result (fp32)
  __shared__ _Float16 pw_hi[2 * kC * kSH];      // 18432 B
  __shared__ _Float16 pw_lo[2 * kC * kSH];      // 18432 B
  __shared__ float    dw_s[2 * kC * 9];         //  4608 B
  __shared__ float    sc_s[4 * kC];             //  1024 B   -> total ~141 KB
  _Float16* t2_hi = (_Float16*)fm_s;            // depthwise2 out reuses fm_s
  _Float16* t2_lo = t2_hi + kOPX * kSH;

  const int tid  = threadIdx.x;
  const int wave = tid >> 5;
  const int lane = tid & 31;
  const int hlf  = lane >> 4;
  const int lm   = lane & 15;

  const int x0 = blockIdx.x * kTX;
  const int y0 = blockIdx.y * kTY;
  const long imgbase = (long)blockIdx.z * H * W;
  const float bninv = 0.9999950000374997f;   // 1/sqrt(1 + 1e-5)

  // ---- stage pointwise weights as split f16 hi/lo ----
  for (int i = tid; i < 2 * kC * 16; i += 256) {
    int l = i / (kC * 16);
    int r = i - l * (kC * 16);
    int n = r >> 4, k4 = r & 15;
    const float* src = l ? pw2 : pw1;
    float4 w = *(const float4*)(src + n * kC + k4 * 4);
    v4h hi, lo;
    split4(w, hi, lo);
    *(v4h*)(&pw_hi[l * kC * kSH + n * kSH + k4 * 4]) = hi;
    *(v4h*)(&pw_lo[l * kC * kSH + n * kSH + k4 * 4]) = lo;
  }
  for (int i = tid; i < 2 * kC * 9; i += 256) {
    int l = i / (kC * 9);
    int r = i - l * (kC * 9);
    dw_s[i] = l ? dw2[r] : dw1[r];
  }
  if (tid < kC) {
    sc_s[tid]          = g1[tid] * bninv;
    sc_s[kC + tid]     = be1[tid];
    sc_s[2 * kC + tid] = g2[tid] * bninv;
    sc_s[3 * kC + tid] = be2[tid];
  }

  // ---- load fm halo tile (zero pad at image borders) ----
  for (int i = tid; i < kFMH * kFMW * 16; i += 256) {
    int p = i >> 4, k4 = i & 15;
    int fy = p / kFMW, fx = p - fy * kFMW;
    int gy = y0 - 2 + fy, gx = x0 - 2 + fx;
    float4 v = {0.f, 0.f, 0.f, 0.f};
    if (gy >= 0 && gy < H && gx >= 0 && gx < W)
      v = *(const float4*)(fm + (imgbase + (long)gy * W + gx) * kC + k4 * 4);
    *(float4*)(&fm_s[p * kC + k4 * 4]) = v;
  }
  __syncthreads();

  // ---- depthwise 3x3 #1 (fp32) -> split f16 t_hi/t_lo (padded rows zero) ----
  for (int i = tid; i < kRROWS * kC; i += 256) {
    int p = i >> 6, c = i & 63;
    float s = 0.f;
    if (p < kRPX) {
      int ry = p / kRW, rx = p - ry * kRW;
      const float* wk = &dw_s[c * 9];
      #pragma unroll
      for (int dy = 0; dy < 3; ++dy)
        #pragma unroll
        for (int dx = 0; dx < 3; ++dx)
          s += wk[dy * 3 + dx] * fm_s[((ry + dy) * kFMW + (rx + dx)) * kC + c];
    }
    _Float16 hi = (_Float16)s;
    t_hi[p * kSH + c] = hi;
    t_lo[p * kSH + c] = (_Float16)(s - (float)hi);
  }
  __syncthreads();

  const v8f vzero = {0.f, 0.f, 0.f, 0.f, 0.f, 0.f, 0.f, 0.f};

  // ---- pointwise1 (split-f16 WMMA) + BN + relu6 + residual -> r_s ----
  for (int mt = wave; mt < kRROWS / 16; mt += 8) {
    const int m0 = mt * 16;
    v8f acc[4];
    #pragma unroll
    for (int nb = 0; nb < 4; ++nb) acc[nb] = vzero;
    #pragma unroll
    for (int kb = 0; kb < kC; kb += 32) {
      const _Float16* pah = &t_hi[(m0 + lm) * kSH + kb + 8 * hlf];
      const _Float16* pal = &t_lo[(m0 + lm) * kSH + kb + 8 * hlf];
      v16h ah = ldfrag2(pah, pah + 16);
      v16h al = ldfrag2(pal, pal + 16);
      #pragma unroll
      for (int nb = 0; nb < 4; ++nb) {
        const int n = nb * 16 + lm;
        const _Float16* pbh = &pw_hi[n * kSH + kb + 8 * hlf];
        const _Float16* pbl = &pw_lo[n * kSH + kb + 8 * hlf];
        v16h bh  = ldfrag2(pbh, pbh + 16);
        v16h blo = ldfrag2(pbl, pbl + 16);
        acc[nb] = wmma32(ah, bh,  acc[nb]);
        acc[nb] = wmma32(ah, blo, acc[nb]);
        acc[nb] = wmma32(al, bh,  acc[nb]);
      }
    }
    #pragma unroll
    for (int nb = 0; nb < 4; ++nb) {
      const int n = nb * 16 + lm;
      #pragma unroll
      for (int j = 0; j < 8; ++j) {
        const int p  = m0 + j + 8 * hlf;
        const int ry = p / kRW, rx = p - ry * kRW;
        const int gy = y0 - 1 + ry, gx = x0 - 1 + rx;
        float v = 0.f;
        // conv2's 'SAME' pad: r outside the image must be exactly zero
        if (p < kRPX && gy >= 0 && gy < H && gx >= 0 && gx < W) {
          v = relu6f(acc[nb][j] * sc_s[n] + sc_s[kC + n]);
          v += fm_s[((ry + 1) * kFMW + (rx + 1)) * kC + n];   // residual
        }
        r_s[p * kC + n] = v;
      }
    }
  }
  __syncthreads();

  // ---- depthwise 3x3 #2 (fp32) -> split f16 t2 (reuses fm_s) ----
  for (int i = tid; i < kOPX * kC; i += 256) {
    int p = i >> 6, c = i & 63;
    int oy = p >> 4, ox = p & 15;
    float s = 0.f;
    const float* wk = &dw_s[kC * 9 + c * 9];
    #pragma unroll
    for (int dy = 0; dy < 3; ++dy)
      #pragma unroll
      for (int dx = 0; dx < 3; ++dx)
        s += wk[dy * 3 + dx] * r_s[((oy + dy) * kRW + (ox + dx)) * kC + c];
    _Float16 hi = (_Float16)s;
    t2_hi[p * kSH + c] = hi;
    t2_lo[p * kSH + c] = (_Float16)(s - (float)hi);
  }
  __syncthreads();

  // ---- pointwise2 (split-f16 WMMA) + BN + relu6 -> global out ----
  for (int mt = wave; mt < kOPX / 16; mt += 8) {
    const int m0 = mt * 16;
    v8f acc[4];
    #pragma unroll
    for (int nb = 0; nb < 4; ++nb) acc[nb] = vzero;
    #pragma unroll
    for (int kb = 0; kb < kC; kb += 32) {
      const _Float16* pah = &t2_hi[(m0 + lm) * kSH + kb + 8 * hlf];
      const _Float16* pal = &t2_lo[(m0 + lm) * kSH + kb + 8 * hlf];
      v16h ah = ldfrag2(pah, pah + 16);
      v16h al = ldfrag2(pal, pal + 16);
      #pragma unroll
      for (int nb = 0; nb < 4; ++nb) {
        const int n = nb * 16 + lm;
        const _Float16* pbh = &pw_hi[kC * kSH + n * kSH + kb + 8 * hlf];
        const _Float16* pbl = &pw_lo[kC * kSH + n * kSH + kb + 8 * hlf];
        v16h bh  = ldfrag2(pbh, pbh + 16);
        v16h blo = ldfrag2(pbl, pbl + 16);
        acc[nb] = wmma32(ah, bh,  acc[nb]);
        acc[nb] = wmma32(ah, blo, acc[nb]);
        acc[nb] = wmma32(al, bh,  acc[nb]);
      }
    }
    #pragma unroll
    for (int nb = 0; nb < 4; ++nb) {
      const int n = nb * 16 + lm;
      #pragma unroll
      for (int j = 0; j < 8; ++j) {
        const int p  = m0 + j + 8 * hlf;
        const int oy = p >> 4, ox = p & 15;
        const int gy = y0 + oy, gx = x0 + ox;
        out[(imgbase + (long)gy * W + gx) * kC + n] =
            relu6f(acc[nb][j] * sc_s[2 * kC + n] + sc_s[3 * kC + n]);
      }
    }
  }
}

// ---------------------------------------------------------------------------
// Host launch
// ---------------------------------------------------------------------------
extern "C" void kernel_launch(void* const* d_in, const int* in_sizes, int n_in,
                              void* d_out, int out_size, void* d_ws, size_t ws_size,
                              hipStream_t stream) {
  (void)n_in; (void)out_size; (void)ws_size;
  const float* feature = (const float*)d_in[0];
  const int*   mask    = (const int*)d_in[1];
  const float* w_in    = (const float*)d_in[2];
  const float* b_in    = (const float*)d_in[3];
  const float* ws_mid  = (const float*)d_in[4];
  const float* bs_mid  = (const float*)d_in[5];
  const float* w_out   = (const float*)d_in[6];
  const float* b_out   = (const float*)d_in[7];
  const float* dw1     = (const float*)d_in[8];
  const float* pw1     = (const float*)d_in[9];
  const float* g1      = (const float*)d_in[10];
  const float* be1     = (const float*)d_in[11];
  const float* dw2     = (const float*)d_in[12];
  const float* pw2     = (const float*)d_in[13];
  const float* g2      = (const float*)d_in[14];
  const float* be2     = (const float*)d_in[15];

  const int N = in_sizes[0] / kC;          // B*H*W
  const int H = 512, W = 512;              // reference-fixed spatial dims
  const int B = N / (H * W);

  float* fm = (float*)d_ws;                // N*64 f32 intermediate (256 MB)

  mlp_kernel<<<N / 128, 256, 0, stream>>>(feature, mask, w_in, b_in, ws_mid, bs_mid,
                                          w_out, b_out, fm);

  dim3 grid(W / kTX, H / kTY, B);
  conv_kernel<<<grid, 256, 0, stream>>>(fm, dw1, pw1, g1, be1, dw2, pw2, g2, be2,
                                        (float*)d_out, H, W);
}
